// DenseR_no_fusion_28424093565773
// MI455X (gfx1250) — compile-verified, matrix operand order checked by reading, not run on hardware
//
#include <hip/hip_runtime.h>
#include <hip/hip_bf16.h>

// ---------------------------------------------------------------------------
// RGCN (3 layers, basis-decomposed) for MI455X / gfx1250.
// GEMMs use v_wmma_f32_16x16x32_bf16 (bf16 in, f32 accumulate).
// ---------------------------------------------------------------------------

typedef __attribute__((ext_vector_type(16))) __bf16 v16bf;
typedef __attribute__((ext_vector_type(8)))  float  v8f;

constexpr int NN = 50000;   // nodes
constexpr int EE = 500000;  // edges
constexpr int RR = 8;       // relations
constexpr int CC = 128;     // channels
constexpr int BB = 4;       // bases

__device__ __forceinline__ unsigned short f2bf(float f) {
  unsigned u = __float_as_uint(f);
  u += 0x7FFFu + ((u >> 16) & 1u);          // round-to-nearest-even
  return (unsigned short)(u >> 16);
}

// ------------------------------ utility ------------------------------------

__global__ void zero_f4(float4* __restrict__ p, long long n4) {
  long long i = (long long)blockIdx.x * blockDim.x + threadIdx.x;
  if (i < n4) p[i] = make_float4(0.f, 0.f, 0.f, 0.f);
}

// counts[r*NN + d] = #edges of relation r into node d  (float, exact < 2^24)
__global__ void count_k(const int* __restrict__ dst, const int* __restrict__ et,
                        float* __restrict__ cnt) {
  int e = blockIdx.x * blockDim.x + threadIdx.x;
  if (e < EE) atomicAdd(&cnt[(size_t)et[e] * NN + dst[e]], 1.0f);
}

__global__ void inv_k(float* __restrict__ cnt) {
  size_t i = (size_t)blockIdx.x * blockDim.x + threadIdx.x;
  if (i < (size_t)RR * NN) cnt[i] = 1.0f / fmaxf(cnt[i], 1.0f);
}

// ------------------------- layer 1: basis combine --------------------------
// Hr[n,c] = sum_b comp1[rel,b] * basis1[b,n,c]   (float4 vectorized)
__global__ void comb_first(const float4* __restrict__ basis, const float* __restrict__ comp1,
                           float4* __restrict__ Hr, int rel) {
  const size_t n4 = (size_t)NN * (CC / 4);
  size_t i = (size_t)blockIdx.x * blockDim.x + threadIdx.x;
  if (i >= n4) return;
  float c0 = comp1[rel * BB + 0], c1 = comp1[rel * BB + 1];
  float c2 = comp1[rel * BB + 2], c3 = comp1[rel * BB + 3];
  float4 b0 = basis[i], b1 = basis[i + n4], b2 = basis[i + 2 * n4], b3 = basis[i + 3 * n4];
  float4 o;
  o.x = c0 * b0.x + c1 * b1.x + c2 * b2.x + c3 * b3.x;
  o.y = c0 * b0.y + c1 * b1.y + c2 * b2.y + c3 * b3.y;
  o.z = c0 * b0.z + c1 * b1.z + c2 * b2.z + c3 * b3.z;
  o.w = c0 * b0.w + c1 * b1.w + c2 * b2.w + c3 * b3.w;
  Hr[i] = o;
}

// -------------------- combined + transposed weights ------------------------
// Wt[r][n][k] = sum_b comp[r,b]*basis[b,k,n]  (r<8) ; Wt[8][n][k] = root[k,n]
// (column-major so B fragments are contiguous k-runs, like A fragments)
__global__ void build_wt(const float* __restrict__ basis, const float* __restrict__ comp,
                         const float* __restrict__ root, unsigned short* __restrict__ Wt,
                         int K) {
  size_t i = (size_t)blockIdx.x * blockDim.x + threadIdx.x;
  size_t tot = (size_t)9 * CC * K;
  if (i >= tot) return;
  int    k = (int)(i % K);
  size_t t = i / K;
  int    n = (int)(t % CC);
  int    r = (int)(t / CC);
  float val;
  if (r == RR) {
    val = root[(size_t)k * CC + n];
  } else {
    float v = 0.f;
#pragma unroll
    for (int b = 0; b < BB; ++b)
      v += comp[r * BB + b] * basis[((size_t)b * K + k) * CC + n];
    val = v;
  }
  Wt[i] = f2bf(val);
}

// ------------------------- f32 -> bf16 activations -------------------------
// Xb[n,k] = bf16(h[n*384 + k]) for k < ncols (layer2: 128, layer3: 256)
__global__ void conv_bf16(const float* __restrict__ h, unsigned short* __restrict__ Xb,
                          int ncols) {
  size_t i = (size_t)blockIdx.x * blockDim.x + threadIdx.x;
  if (i >= (size_t)NN * ncols) return;
  int    k = (int)(i % ncols);
  size_t n = i / ncols;
  Xb[i] = f2bf(h[n * 384 + k]);
}

// ------------------------------ WMMA GEMM ----------------------------------
// One wave computes a 16x128 strip of Hr = Xb @ Wr (bf16 inputs, f32 acc).
// A fragment (16x32 bf16): lane L<16 holds row M=L, K {kk..kk+7, kk+16..kk+23};
// lanes 16-31 hold K {kk+8..kk+15, kk+24..kk+31} -> two contiguous 16B loads.
// B fragment mirrors A with columns, enabled by the transposed Wt layout.
union FragAB { uint4 q[2]; v16bf v; };

__global__ void __launch_bounds__(256) gemm_bf16_k(
    const unsigned short* __restrict__ Xb,  // [NN, K]
    const unsigned short* __restrict__ Wt,  // [9, 128, K]
    float* __restrict__ Hr,                 // [NN, 128]
    int K, int rel) {
  const int wave  = threadIdx.x >> 5;
  const int lane  = threadIdx.x & 31;
  const int mtile = blockIdx.x * 8 + wave;
  if (mtile * 16 >= NN) return;              // wave-uniform exit (EXEC stays full)
  const int lm = lane & 15;
  const int lh = lane >> 4;

  const unsigned short* arow  = Xb + (size_t)(mtile * 16 + lm) * K + lh * 8;
  const unsigned short* wbase = Wt + (size_t)rel * CC * K + lh * 8;

  v8f acc[8] = {};
  for (int kk = 0; kk < K; kk += 32) {
    FragAB a;
    a.q[0] = *(const uint4*)(arow + kk);
    a.q[1] = *(const uint4*)(arow + kk + 16);
#pragma unroll
    for (int nt = 0; nt < 8; ++nt) {
      const unsigned short* brow = wbase + (size_t)(nt * 16 + lm) * K;
      FragAB b;
      b.q[0] = *(const uint4*)(brow + kk);
      b.q[1] = *(const uint4*)(brow + kk + 16);
      acc[nt] = __builtin_amdgcn_wmma_f32_16x16x32_bf16(
          false, a.v, false, b.v, (short)0, acc[nt], false, false);
    }
  }
  // D layout: lane -> col nt*16+lm ; vgpr v -> row lh*8+v
  const int row0 = mtile * 16 + lh * 8;
#pragma unroll
  for (int nt = 0; nt < 8; ++nt) {
    float* out = Hr + (size_t)row0 * CC + nt * 16 + lm;
#pragma unroll
    for (int v = 0; v < 8; ++v) out[(size_t)v * CC] = acc[nt][v];
  }
}

// ------------------------------ edge scatter -------------------------------
// One wave per edge: agg[dst,:] += Hr[src,:] * inv_cnt[rel,dst]   (128 ch)
__global__ void __launch_bounds__(256) scatter_k(
    const float* __restrict__ Hr, float* __restrict__ agg,
    const int* __restrict__ src, const int* __restrict__ dst,
    const int* __restrict__ et, const float* __restrict__ invc, int rel) {
  int e = blockIdx.x * 8 + (threadIdx.x >> 5);
  if (e >= EE) return;
  if (et[e] != rel) return;
  int lane = threadIdx.x & 31;
  int s = src[e], d = dst[e];
  float  w = invc[(size_t)rel * NN + d];
  float4 v = *(const float4*)(Hr + (size_t)s * CC + lane * 4);
  float* a = agg + (size_t)d * CC + lane * 4;
  atomicAdd(a + 0, v.x * w);
  atomicAdd(a + 1, v.y * w);
  atomicAdd(a + 2, v.z * w);
  atomicAdd(a + 3, v.w * w);
}

// ------------------------------- epilogue ----------------------------------
// h[n, coloff+c] = relu(agg[n,c] + rootterm[n,c] + bias[c])
__global__ void epilogue_k(const float* __restrict__ agg, const float* __restrict__ rt,
                           const float* __restrict__ bias, float* __restrict__ h,
                           int coloff) {
  size_t i = (size_t)blockIdx.x * blockDim.x + threadIdx.x;
  if (i >= (size_t)NN * CC) return;
  int    c = (int)(i & (CC - 1));
  size_t n = i >> 7;
  float  v = agg[i] + rt[i] + bias[c];
  h[n * 384 + coloff + c] = fmaxf(v, 0.0f);
}

// ------------------------------ log-softmax --------------------------------
// In place over 384 channels; one 128-thread block per row, 3 values/thread.
__global__ void __launch_bounds__(128) logsoftmax_k(float* __restrict__ h) {
  const int n = blockIdx.x;
  const int t = threadIdx.x;
  float* row = h + (size_t)n * 384;
  float v0 = row[t], v1 = row[t + 128], v2 = row[t + 256];
  __shared__ float red[128];
  float m = fmaxf(v0, fmaxf(v1, v2));
  red[t] = m;
  __syncthreads();
  for (int s = 64; s > 0; s >>= 1) {
    if (t < s) red[t] = fmaxf(red[t], red[t + s]);
    __syncthreads();
  }
  m = red[0];
  __syncthreads();
  float ssum = expf(v0 - m) + expf(v1 - m) + expf(v2 - m);
  red[t] = ssum;
  __syncthreads();
  for (int s = 64; s > 0; s >>= 1) {
    if (t < s) red[t] += red[t + s];
    __syncthreads();
  }
  float lse = m + logf(red[0]);
  row[t]       = v0 - lse;
  row[t + 128] = v1 - lse;
  row[t + 256] = v2 - lse;
}

// ------------------------------ host driver --------------------------------

static inline int cdiv(long long a, long long b) { return (int)((a + b - 1) / b); }

extern "C" void kernel_launch(void* const* d_in, const int* in_sizes, int n_in,
                              void* d_out, int out_size, void* d_ws, size_t ws_size,
                              hipStream_t stream) {
  (void)in_sizes; (void)n_in; (void)out_size; (void)ws_size;

  const int*   edge_index = (const int*)d_in[0];   // [2,E]
  const int*   edge_type  = (const int*)d_in[1];   // [E]
  const float* basis1     = (const float*)d_in[2]; // [B,N,C]
  const float* comp1      = (const float*)d_in[3]; // [R,B]
  const float* root1      = (const float*)d_in[4]; // [N,C]
  const float* bias1      = (const float*)d_in[5]; // [C]
  const float* basis_b0   = (const float*)d_in[6]; // [B,C,C]
  const float* comp_b0    = (const float*)d_in[7];
  const float* root_b0    = (const float*)d_in[8]; // [C,C]
  const float* bias_b0    = (const float*)d_in[9];
  const float* basis_b1   = (const float*)d_in[10]; // [B,2C,C]
  const float* comp_b1    = (const float*)d_in[11];
  const float* root_b1    = (const float*)d_in[12]; // [2C,C]
  const float* bias_b1    = (const float*)d_in[13];

  const int* src = edge_index;
  const int* dst = edge_index + EE;

  float* h = (float*)d_out;                        // [N,384] built in place

  // workspace layout (all offsets 256B aligned); total ~79 MB
  char* ws = (char*)d_ws;
  float*          agg  = (float*)(ws);                       // N*128 f32  = 25.6 MB
  float*          Hr   = (float*)(ws + 25600000);            // N*128 f32  = 25.6 MB
  unsigned short* Xb   = (unsigned short*)(ws + 51200000);   // N*256 bf16 = 25.6 MB
  unsigned short* Wt   = (unsigned short*)(ws + 76800000);   // 9*128*256 bf16
  float*          invc = (float*)(ws + 77389824);            // R*N f32    = 1.6 MB

  const long long nc4    = (long long)NN * CC / 4;   // float4 count of [N,128]
  const int gemm_blocks  = cdiv(NN / 16, 8);         // 3125 M-tiles, 8 waves/block
  const int scat_blocks  = cdiv(EE, 8);              // 1 wave per edge

  // ---- per-(relation,dst) inverse edge counts (shared by all 3 layers) ----
  zero_f4<<<cdiv((long long)RR * NN / 4, 256), 256, 0, stream>>>((float4*)invc, (long long)RR * NN / 4);
  count_k<<<cdiv(EE, 256), 256, 0, stream>>>(dst, edge_type, invc);
  inv_k<<<cdiv((long long)RR * NN, 256), 256, 0, stream>>>(invc);

  // ------------------------------ layer 1 ---------------------------------
  zero_f4<<<cdiv(nc4, 256), 256, 0, stream>>>((float4*)agg, nc4);
  for (int r = 0; r < RR; ++r) {
    comb_first<<<cdiv(nc4, 256), 256, 0, stream>>>((const float4*)basis1, comp1, (float4*)Hr, r);
    scatter_k<<<scat_blocks, 256, 0, stream>>>(Hr, agg, src, dst, edge_type, invc, r);
  }
  epilogue_k<<<cdiv((long long)NN * CC, 256), 256, 0, stream>>>(agg, root1, bias1, h, 0);

  // ------------------------------ layer 2 (K=128) -------------------------
  conv_bf16<<<cdiv((long long)NN * 128, 256), 256, 0, stream>>>(h, Xb, 128);
  build_wt<<<cdiv((long long)9 * CC * 128, 256), 256, 0, stream>>>(basis_b0, comp_b0, root_b0, Wt, 128);
  zero_f4<<<cdiv(nc4, 256), 256, 0, stream>>>((float4*)agg, nc4);
  for (int r = 0; r < RR; ++r) {
    gemm_bf16_k<<<gemm_blocks, 256, 0, stream>>>(Xb, Wt, Hr, 128, r);
    scatter_k<<<scat_blocks, 256, 0, stream>>>(Hr, agg, src, dst, edge_type, invc, r);
  }
  gemm_bf16_k<<<gemm_blocks, 256, 0, stream>>>(Xb, Wt, Hr, 128, RR);  // x @ root
  epilogue_k<<<cdiv((long long)NN * CC, 256), 256, 0, stream>>>(agg, Hr, bias_b0, h, 128);

  // ------------------------------ layer 3 (K=256) -------------------------
  conv_bf16<<<cdiv((long long)NN * 256, 256), 256, 0, stream>>>(h, Xb, 256);
  build_wt<<<cdiv((long long)9 * CC * 256, 256), 256, 0, stream>>>(basis_b1, comp_b1, root_b1, Wt, 256);
  zero_f4<<<cdiv(nc4, 256), 256, 0, stream>>>((float4*)agg, nc4);
  for (int r = 0; r < RR; ++r) {
    gemm_bf16_k<<<gemm_blocks, 256, 0, stream>>>(Xb, Wt, Hr, 256, r);
    scatter_k<<<scat_blocks, 256, 0, stream>>>(Hr, agg, src, dst, edge_type, invc, r);
  }
  gemm_bf16_k<<<gemm_blocks, 256, 0, stream>>>(Xb, Wt, Hr, 256, RR);  // x @ root
  epilogue_k<<<cdiv((long long)NN * CC, 256), 256, 0, stream>>>(agg, Hr, bias_b1, h, 256);

  // ------------------------------ log-softmax -----------------------------
  logsoftmax_k<<<NN, 128, 0, stream>>>(h);
}
